// VectorQuantize_82523501625771
// MI455X (gfx1250) — compile-verified, matrix operand order checked by reading, not run on hardware
//
#include <hip/hip_runtime.h>
#include <hip/hip_bf16.h>

typedef _Float16 v16h __attribute__((ext_vector_type(16)));
typedef _Float16 v8h  __attribute__((ext_vector_type(8)));
typedef float    v8f  __attribute__((ext_vector_type(8)));

#define VQ_DIM 256
#define VQ_NE  8192
#define VQ_NQ  8192
#define VQ_CHUNK  2048                 // codes per block
#define VQ_NCHUNK (VQ_NE / VQ_CHUNK)   // 4
#define VQ_BETA 0.25f

// ---------------------------------------------------------------------------
// Kernel A: row-wise L2 normalize (f32 in) -> f16 out. One wave per row.
// ---------------------------------------------------------------------------
__global__ void vq_normalize_f16(const float* __restrict__ src,
                                 _Float16* __restrict__ dst) {
    const int wave = threadIdx.x >> 5;
    const int lane = threadIdx.x & 31;
    const int row  = blockIdx.x * 8 + wave;

    const float* p = src + (size_t)row * VQ_DIM + lane * 8;
    float4 v0 = *(const float4*)p;
    float4 v1 = *(const float4*)(p + 4);

    float s = v0.x*v0.x + v0.y*v0.y + v0.z*v0.z + v0.w*v0.w
            + v1.x*v1.x + v1.y*v1.y + v1.z*v1.z + v1.w*v1.w;
#pragma unroll
    for (int m = 16; m; m >>= 1) s += __shfl_xor(s, m, 32);

    const float scale = 1.0f / fmaxf(sqrtf(s), 1e-12f);

    v8h h;
    h[0] = (_Float16)(v0.x * scale); h[1] = (_Float16)(v0.y * scale);
    h[2] = (_Float16)(v0.z * scale); h[3] = (_Float16)(v0.w * scale);
    h[4] = (_Float16)(v1.x * scale); h[5] = (_Float16)(v1.y * scale);
    h[6] = (_Float16)(v1.z * scale); h[7] = (_Float16)(v1.w * scale);
    *(v8h*)(dst + (size_t)row * VQ_DIM + lane * 8) = h;
}

// Build a 16x32 f16 WMMA fragment per the CDNA5 A/B register layout:
// this lane's two contiguous 8-half chunks sit at +0 and +16 halves.
__device__ __forceinline__ v16h vq_load_frag(const _Float16* base) {
    v8h lo = *(const v8h*)base;
    v8h hi = *(const v8h*)(base + 16);
    v16h r;
#pragma unroll
    for (int i = 0; i < 8; ++i) { r[i] = lo[i]; r[i + 8] = hi[i]; }
    return r;
}

// ---------------------------------------------------------------------------
// Kernel B: similarity GEMM + partial argmax.
// Wave = 16 queries vs one 2048-code chunk.  Per 16-code tile:
//   16x global_load_b128 (one clause)  |sched_barrier|  one wait  ->
//   8 back-to-back v_wmma_f32_16x16x32_f16,
// while global_prefetch warms WGP$ one tile ahead at zero VGPR cost.
// waves_per_eu(1) lifts the default occupancy-driven ~128-VGPR cap so the
// A tile (64 regs) and B tile (64 regs) both stay resident (no spills).
// Grid: 64 query-blocks x 4 code-chunks = 256 blocks.
// ---------------------------------------------------------------------------
__global__ void
__launch_bounds__(256)
__attribute__((amdgpu_waves_per_eu(1)))
vq_argmax(const _Float16* __restrict__ zh,
          const _Float16* __restrict__ eh,
          float* __restrict__ pbest,
          int* __restrict__ pidx) {
    const int qb    = blockIdx.x & 63;
    const int chunk = blockIdx.x >> 6;
    const int wave  = threadIdx.x >> 5;
    const int lane  = threadIdx.x & 31;
    const int m0    = qb * 128 + wave * 16;
    const int half  = lane >> 4;
    const int l16   = lane & 15;
    const int koff  = half * 8;
    const int nbase = chunk * VQ_CHUNK;
    const int nend  = nbase + VQ_CHUNK;

    // Preload all 8 A fragments (the wave's 16 query rows, full K=256).
    v16h a[8];
    const _Float16* arow = zh + (size_t)(m0 + l16) * VQ_DIM;
#pragma unroll
    for (int f = 0; f < 8; ++f) a[f] = vq_load_frag(arow + f * 32 + koff);

    float best[8];
    int   bidx[8];
#pragma unroll
    for (int r = 0; r < 8; ++r) { best[r] = -1e30f; bidx[r] = 0; }

    // This lane's pointer into the first B tile (row nbase + l16).
    const _Float16* bptr = eh + (size_t)(nbase + l16) * VQ_DIM + koff;

    for (int n0 = nbase; n0 < nend; n0 += 16) {
        // Whole B tile (16 independent b128 loads -> one clause, one wait).
        v16h b[8];
#pragma unroll
        for (int f = 0; f < 8; ++f) b[f] = vq_load_frag(bptr + f * 32);

        // Warm WGP$ for the next tile (no VGPRs, no counters).
        if (n0 + 16 < nend) {                         // scalar-uniform guard
            const char* np = (const char*)(bptr + 16 * VQ_DIM - koff);
            __builtin_prefetch(np, 0, 3);
            __builtin_prefetch(np + 128, 0, 3);
            __builtin_prefetch(np + 256, 0, 3);
            __builtin_prefetch(np + 384, 0, 3);
        }

        // Keep the scheduler from interleaving loads with the WMMA burst:
        // everything above issues first, then 8 chained WMMAs.
        __builtin_amdgcn_sched_barrier(0);

        v8f c = {};
#pragma unroll
        for (int f = 0; f < 8; ++f) {
            c = __builtin_amdgcn_wmma_f32_16x16x32_f16(
                    false, a[f], false, b[f], (short)0, c, false, false);
        }

        const int n = n0 + l16;
#pragma unroll
        for (int r = 0; r < 8; ++r) {
            if (c[r] > best[r]) { best[r] = c[r]; bidx[r] = n; }
        }

        bptr += 16 * VQ_DIM;
    }

    // Argmax across the 16 lanes of each half-wave (C: N = lane&15).
#pragma unroll
    for (int r = 0; r < 8; ++r) {
#pragma unroll
        for (int m = 8; m; m >>= 1) {
            float ov = __shfl_xor(best[r], m, 16);
            int   oi = __shfl_xor(bidx[r], m, 16);
            if (ov > best[r] || (ov == best[r] && oi < bidx[r])) {
                best[r] = ov; bidx[r] = oi;
            }
        }
    }

    if (l16 == 0) {
        const int mbase = chunk * VQ_NQ + m0 + half * 8;   // C: M = r + 8*half
#pragma unroll
        for (int r = 0; r < 8; ++r) {
            pbest[mbase + r] = best[r];
            pidx [mbase + r] = bidx[r];
        }
    }
}

// ---------------------------------------------------------------------------
// Kernel B2: merge the per-chunk partial argmax results (ascending chunk
// order => first-occurrence tie-breaking like jnp.argmin).
// ---------------------------------------------------------------------------
__global__ void vq_combine(const float* __restrict__ pbest,
                           const int* __restrict__ pidx,
                           int* __restrict__ idxOut) {
    const int q = blockIdx.x * 256 + threadIdx.x;
    float bv = pbest[q];
    int   bi = pidx[q];
#pragma unroll
    for (int c = 1; c < VQ_NCHUNK; ++c) {
        float v = pbest[c * VQ_NQ + q];
        int   i = pidx [c * VQ_NQ + q];
        if (v > bv || (v == bv && i < bi)) { bv = v; bi = i; }
    }
    idxOut[q] = bi;
}

// ---------------------------------------------------------------------------
// Kernel C: gather code row, renormalize in f32, emit z_q + per-row loss.
// One wave per row.
// ---------------------------------------------------------------------------
__global__ void vq_output(const float* __restrict__ z,
                          const float* __restrict__ E,
                          const int* __restrict__ idxArr,
                          float* __restrict__ zq_out,
                          float* __restrict__ idx_out_f,
                          float* __restrict__ rowloss) {
    const int wave = threadIdx.x >> 5;
    const int lane = threadIdx.x & 31;
    const int row  = blockIdx.x * 8 + wave;
    const int idx  = idxArr[row];

    const float* ep = E + (size_t)idx * VQ_DIM + lane * 8;
    float4 e0 = *(const float4*)ep;
    float4 e1 = *(const float4*)(ep + 4);
    float se = e0.x*e0.x + e0.y*e0.y + e0.z*e0.z + e0.w*e0.w
             + e1.x*e1.x + e1.y*e1.y + e1.z*e1.z + e1.w*e1.w;
#pragma unroll
    for (int m = 16; m; m >>= 1) se += __shfl_xor(se, m, 32);
    const float escale = 1.0f / fmaxf(sqrtf(se), 1e-12f);

    const float* zp = z + (size_t)row * VQ_DIM + lane * 8;
    float4 z0 = *(const float4*)zp;
    float4 z1 = *(const float4*)(zp + 4);
    float sz = z0.x*z0.x + z0.y*z0.y + z0.z*z0.z + z0.w*z0.w
             + z1.x*z1.x + z1.y*z1.y + z1.z*z1.z + z1.w*z1.w;
#pragma unroll
    for (int m = 16; m; m >>= 1) sz += __shfl_xor(sz, m, 32);
    const float zscale = 1.0f / fmaxf(sqrtf(sz), 1e-12f);

    float4 q0, q1;
    q0.x = e0.x * escale; q0.y = e0.y * escale;
    q0.z = e0.z * escale; q0.w = e0.w * escale;
    q1.x = e1.x * escale; q1.y = e1.y * escale;
    q1.z = e1.z * escale; q1.w = e1.w * escale;

    float d, diff = 0.0f;
    d = q0.x - z0.x * zscale; diff += d * d;
    d = q0.y - z0.y * zscale; diff += d * d;
    d = q0.z - z0.z * zscale; diff += d * d;
    d = q0.w - z0.w * zscale; diff += d * d;
    d = q1.x - z1.x * zscale; diff += d * d;
    d = q1.y - z1.y * zscale; diff += d * d;
    d = q1.z - z1.z * zscale; diff += d * d;
    d = q1.w - z1.w * zscale; diff += d * d;

    float* op = zq_out + (size_t)row * VQ_DIM + lane * 8;
    *(float4*)op       = q0;
    *(float4*)(op + 4) = q1;

#pragma unroll
    for (int m = 16; m; m >>= 1) diff += __shfl_xor(diff, m, 32);
    if (lane == 0) {
        rowloss[row]   = diff;
        idx_out_f[row] = (float)idx;
    }
}

// ---------------------------------------------------------------------------
// Kernel D: deterministic single-block loss reduction (fixed order).
// ---------------------------------------------------------------------------
__global__ void vq_loss(const float* __restrict__ rowloss,
                        float* __restrict__ loss) {
    __shared__ float sm[256];
    float s = 0.0f;
#pragma unroll
    for (int j = 0; j < VQ_NQ / 256; ++j) s += rowloss[threadIdx.x + 256 * j];
    sm[threadIdx.x] = s;
    __syncthreads();
    for (int st = 128; st; st >>= 1) {
        if ((int)threadIdx.x < st) sm[threadIdx.x] += sm[threadIdx.x + st];
        __syncthreads();
    }
    if (threadIdx.x == 0)
        loss[0] = (1.0f + VQ_BETA) * sm[0] / ((float)VQ_NQ * (float)VQ_DIM);
}

// ---------------------------------------------------------------------------
extern "C" void kernel_launch(void* const* d_in, const int* in_sizes, int n_in,
                              void* d_out, int out_size, void* d_ws, size_t ws_size,
                              hipStream_t stream) {
    (void)in_sizes; (void)n_in; (void)out_size; (void)ws_size;

    const float* z = (const float*)d_in[0];          // [8,1024,256] f32
    const float* E = (const float*)d_in[1];          // [8192,256]   f32

    float* out      = (float*)d_out;
    float* zq_out   = out;                           // 8*1024*256
    float* loss_out = out + (size_t)VQ_NQ * VQ_DIM;  // 1
    float* idx_out  = loss_out + 1;                  // 8192 (as float)

    char* ws = (char*)d_ws;
    _Float16* zh      = (_Float16*)(ws);                           // 4 MB
    _Float16* eh      = (_Float16*)(ws + (4u << 20));              // 4 MB
    float*    pbest   = (float*) (ws + (8u << 20));                // 128 KB
    int*      pidx    = (int*)   (ws + (8u << 20) + (128u << 10)); // 128 KB
    int*      idxArr  = (int*)   (ws + (8u << 20) + (256u << 10)); // 32 KB
    float*    rowloss = (float*) (ws + (8u << 20) + (288u << 10)); // 32 KB

    vq_normalize_f16<<<VQ_NQ / 8, 256, 0, stream>>>(z, zh);
    vq_normalize_f16<<<VQ_NE / 8, 256, 0, stream>>>(E, eh);
    vq_argmax      <<<64 * VQ_NCHUNK, 256, 0, stream>>>(zh, eh, pbest, pidx);
    vq_combine     <<<VQ_NQ / 256, 256, 0, stream>>>(pbest, pidx, idxArr);
    vq_output      <<<VQ_NQ / 8, 256, 0, stream>>>(z, E, idxArr, zq_out,
                                                   idx_out, rowloss);
    vq_loss        <<<1, 256, 0, stream>>>(rowloss, loss_out);
}
